// GroupedVectorQuantizer_35648228556983
// MI455X (gfx1250) — compile-verified, matrix operand order checked by reading, not run on hardware
//
#include <hip/hip_runtime.h>
#include <hip/hip_bf16.h>

typedef __attribute__((ext_vector_type(16))) __bf16 v16bf;
typedef __attribute__((ext_vector_type(8)))  float  v8f;
typedef __attribute__((ext_vector_type(4)))  float  vf4;

#define BATCH       4096
#define NUM_G       32
#define KCODES      1024
#define DIM         64
#define NROWS       (NUM_G * KCODES)           // 32768 codebook rows
#define NPAIRS      (BATCH * NUM_G)            // 131072
#define OUT_Q_ELEMS (BATCH * NUM_G * DIM)      // 8388608
#define ZROW        (NUM_G * DIM)              // 2048
#define ROW_H       72                          // LDS row stride in bf16 (144 B, bank-conflict free)

// f32 -> bf16 RNE; returns the 16 bits and the exact f32 the bf16 represents.
__device__ __forceinline__ unsigned bf16_bits_rne(float x, float* rep) {
    unsigned u = __float_as_uint(x);
    unsigned r = (u + 0x7FFFu + ((u >> 16) & 1u)) & 0xFFFF0000u;
    *rep = __uint_as_float(r);
    return r >> 16;
}

__device__ __forceinline__ __bf16 f32_to_bf16_rne(float x, float* rep) {
    unsigned hs = bf16_bits_rne(x, rep);
    return __builtin_bit_cast(__bf16, (unsigned short)hs);
}

// K index for element j of a WMMA bf16 fragment f (0: K=0..31, 1: K=32..63),
// per the CDNA5 ISA 16-bit A-matrix 16x32 VGPR layout.
__device__ __forceinline__ int frag_k(int f, int j, int hihalf) {
    int v = j >> 1, h = j & 1;
    return f * 32 + ((v < 4) ? 0 : 16) + hihalf * 8 + (v & 3) * 2 + h;
}

struct frag_bits { vf4 a, b; };
// One contiguous 32B chunk of a pre-permuted LDS row -> v16bf fragment.
__device__ __forceinline__ v16bf load_frag(const unsigned short* p) {
    frag_bits t;
    t.a = *(const vf4*)(p);
    t.b = *(const vf4*)(p + 8);
    return __builtin_bit_cast(v16bf, t);
}

// ---------------------------------------------------------------------------
// Kernel 0b: codebook preprocessing. One wave per code row.
// Emits cb_hi / cb_lo rows permuted into WMMA B-fragment order:
// chunk(f,hb) = K in {f*32+hb*8 .. +7} U {f*32+16+hb*8 .. +7}  (4 x 32B chunks)
// Also exact f32 c2 per row.
// ---------------------------------------------------------------------------
__global__ __launch_bounds__(256) void vq_prep_kernel(
    const float* __restrict__ cb, unsigned* __restrict__ cbh32,
    unsigned* __restrict__ cbl32, float* __restrict__ c2out)
{
    const int row  = blockIdx.x * 8 + (threadIdx.x >> 5);
    const int lane = threadIdx.x & 31;
    const int p      = lane * 2;              // output position (permuted order)
    const int chunk  = p >> 4;                // 0..3
    const int within = p & 15;
    const int f  = chunk >> 1;
    const int hb = chunk & 1;
    const int K  = f * 32 + hb * 8 + (within & 7) + ((within >> 3) << 4);

    const float* src = cb + (size_t)row * DIM + K;   // K, K+1 consecutive
    float x0 = src[0], x1 = src[1];

    float rh0, rh1, rl0, rl1;
    unsigned h0 = bf16_bits_rne(x0, &rh0);
    unsigned h1 = bf16_bits_rne(x1, &rh1);
    unsigned l0 = bf16_bits_rne(x0 - rh0, &rl0);
    unsigned l1 = bf16_bits_rne(x1 - rh1, &rl1);

    const size_t o = (size_t)row * 32 + lane;        // packed u32 index
    cbh32[o] = h0 | (h1 << 16);
    cbl32[o] = l0 | (l1 << 16);

    float s = x0 * x0 + x1 * x1;
    #pragma unroll
    for (int off = 16; off >= 1; off >>= 1) s += __shfl_xor(s, off, 32);
    if (lane == 0) c2out[row] = s;
}

// ---------------------------------------------------------------------------
// Kernel 1: grouped GEMM (split-bf16 WMMA) + argmin over the 1024 codes.
// grid = (32 m-blocks, 32 groups), block = 256 threads (8 waves, 16 rows/wave)
// ---------------------------------------------------------------------------
__global__ __launch_bounds__(256) void vq_argmin_kernel(
    const float* __restrict__ z,
    const float4* __restrict__ cbh, const float4* __restrict__ cbl,
    const float* __restrict__ c2g,
    int* __restrict__ idx_ws, float* __restrict__ out_idx)
{
    // [buf][hi=0/lo=1][row * ROW_H]  (144B row stride: conflict-free, 16B aligned)
    __shared__ __align__(16) unsigned short lds[2][2][16 * ROW_H];

    const int g      = blockIdx.y;
    const int lane   = threadIdx.x & 31;
    const int wave   = threadIdx.x >> 5;
    const int hihalf = lane >> 4;
    const int nl     = lane & 15;
    const int m_base = blockIdx.x * 128 + wave * 16;

    // ---- A fragments (z rows), hi/lo bf16 split, built once ----
    v16bf a_hi[2], a_lo[2];
    {
        const float* zrow = z + (size_t)(m_base + nl) * ZROW + g * DIM;
        #pragma unroll
        for (int f = 0; f < 2; ++f)
            #pragma unroll
            for (int j = 0; j < 16; ++j) {
                int K = frag_k(f, j, hihalf);
                float x = zrow[K];
                float rhi, rlo;
                __bf16 h = f32_to_bf16_rne(x, &rhi);
                __bf16 l = f32_to_bf16_rne(x - rhi, &rlo);
                a_hi[f][j] = h;
                a_lo[f][j] = l;
            }
    }

    float minval[8];
    int   minidx[8];
    #pragma unroll
    for (int v = 0; v < 8; ++v) { minval[v] = 3.402823466e+38f; minidx[v] = 0; }

    // staging: 256 threads move 16 rows x 128B of hi AND lo per tile.
    // thread -> (which = hi/lo, row r, 16B chunk c)
    const int which = threadIdx.x >> 7;            // 0 = hi, 1 = lo
    const int r_st  = (threadIdx.x >> 3) & 15;     // staged row
    const int c_st  = threadIdx.x & 7;             // 16B chunk within 128B row
    const float4* gsrc = (which == 0) ? cbh : cbl;
    const size_t  grow0 = (size_t)g * KCODES;      // first row of this group
    unsigned short* ldst0 = &lds[0][which][r_st * ROW_H + c_st * 8];
    unsigned short* ldst1 = &lds[1][which][r_st * ROW_H + c_st * 8];
    const float*  c2grp = c2g + (size_t)g * KCODES;

    // stage tile 0
    *(float4*)ldst0 = gsrc[(grow0 + r_st) * 8 + c_st];
    __syncthreads();

    for (int t = 0; t < 64; ++t) {
        const int cur = t & 1;
        float4 pref = make_float4(0.f, 0.f, 0.f, 0.f);
        if (t + 1 < 64)
            pref = gsrc[(grow0 + (t + 1) * 16 + r_st) * 8 + c_st];
        if (t + 2 < 64)
            __builtin_prefetch(&gsrc[(grow0 + (t + 2) * 16 + r_st) * 8 + c_st], 0, 1);

        // exact squared-norm of this lane's column
        const float c2v = c2grp[t * 16 + nl];

        // ---- B fragments: one contiguous 32B LDS chunk each ----
        const int rowoff = nl * ROW_H;
        v16bf bh0 = load_frag(&lds[cur][0][rowoff + (0 * 2 + hihalf) * 16]);
        v16bf bh1 = load_frag(&lds[cur][0][rowoff + (1 * 2 + hihalf) * 16]);
        v16bf bl0 = load_frag(&lds[cur][1][rowoff + (0 * 2 + hihalf) * 16]);
        v16bf bl1 = load_frag(&lds[cur][1][rowoff + (1 * 2 + hihalf) * 16]);

        // ---- split-bf16 WMMA: zc ~= Ah*Bh + Ah*Bl + Al*Bh (f32 accum) ----
        v8f acc = {};
        acc = __builtin_amdgcn_wmma_f32_16x16x32_bf16(false, a_hi[0], false, bh0, (short)0, acc, false, false);
        acc = __builtin_amdgcn_wmma_f32_16x16x32_bf16(false, a_hi[1], false, bh1, (short)0, acc, false, false);
        acc = __builtin_amdgcn_wmma_f32_16x16x32_bf16(false, a_hi[0], false, bl0, (short)0, acc, false, false);
        acc = __builtin_amdgcn_wmma_f32_16x16x32_bf16(false, a_hi[1], false, bl1, (short)0, acc, false, false);
        acc = __builtin_amdgcn_wmma_f32_16x16x32_bf16(false, a_lo[0], false, bh0, (short)0, acc, false, false);
        acc = __builtin_amdgcn_wmma_f32_16x16x32_bf16(false, a_lo[1], false, bh1, (short)0, acc, false, false);

        // score = c2[k] - 2*zc   (z2 per-row constant dropped)
        const int colbase = t * 16 + nl;
        #pragma unroll
        for (int v = 0; v < 8; ++v) {
            float score = c2v - 2.0f * acc[v];
            if (score < minval[v]) { minval[v] = score; minidx[v] = colbase; }
        }

        if (t + 1 < 64)
            *(float4*)((cur == 0) ? ldst1 : ldst0) = pref;
        __syncthreads();
    }

    // ---- cross-lane argmin within each 16-lane half (lowest-index ties) ----
    #pragma unroll
    for (int v = 0; v < 8; ++v) {
        float bv = minval[v];
        int   bi = minidx[v];
        #pragma unroll
        for (int off = 8; off >= 1; off >>= 1) {
            float ov = __shfl_xor(bv, off, 32);
            int   oi = __shfl_xor(bi, off, 32);
            if (ov < bv || (ov == bv && oi < bi)) { bv = ov; bi = oi; }
        }
        if (nl == 0) {
            int rowm = m_base + v + hihalf * 8;
            idx_ws[rowm * NUM_G + g]  = bi;
            out_idx[rowm * NUM_G + g] = (float)bi;
        }
    }
}

// ---------------------------------------------------------------------------
// Kernel 2: gather z_q, squared-error accumulation, usage histogram.
// ---------------------------------------------------------------------------
__global__ __launch_bounds__(256) void vq_gather_kernel(
    const float* __restrict__ z, const float* __restrict__ cb,
    const int* __restrict__ idx_ws, float* __restrict__ out_q,
    float* __restrict__ loss_acc, unsigned* __restrict__ counts)
{
    const int lane64 = threadIdx.x & 63;
    const int sub    = threadIdx.x >> 6;
    const int p      = blockIdx.x * 4 + sub;     // (b,g) pair, p = b*32+g
    const int b      = p >> 5;
    const int g      = p & 31;
    const int k      = idx_ws[p];

    const float  c    = cb[((size_t)(g * KCODES + k)) * DIM + lane64];
    const size_t zoff = (size_t)b * ZROW + g * DIM + lane64;
    const float  zv   = z[zoff];
    out_q[zoff] = c;
    if (lane64 == 0) atomicAdd(&counts[k], 1u);

    float d = zv - c;
    float s = d * d;
    #pragma unroll
    for (int off = 16; off >= 1; off >>= 1) s += __shfl_xor(s, off, 32);

    __shared__ float red[8];
    if ((threadIdx.x & 31) == 0) red[threadIdx.x >> 5] = s;
    __syncthreads();
    if (threadIdx.x == 0) {
        float tot = 0.f;
        #pragma unroll
        for (int i = 0; i < 8; ++i) tot += red[i];
        atomicAdd(loss_acc, tot);
    }
}

// ---------------------------------------------------------------------------
// Kernel 3: entropy / perplexity / losses.
// ---------------------------------------------------------------------------
__global__ __launch_bounds__(1024) void vq_finalize_kernel(
    const unsigned* __restrict__ counts, const float* __restrict__ loss_acc,
    float* __restrict__ out_scalars)
{
    __shared__ float sh[1024];
    const int tid = threadIdx.x;
    float usage = (float)counts[tid] * (1.0f / (float)NPAIRS);
    sh[tid] = -usage * logf(usage + 1e-10f);
    __syncthreads();
    for (int s = 512; s > 0; s >>= 1) {
        if (tid < s) sh[tid] += sh[tid + s];
        __syncthreads();
    }
    if (tid == 0) {
        float entropy = sh[0];
        float loss = loss_acc[0] * (1.0f / (float)OUT_Q_ELEMS);
        out_scalars[0] = loss;
        out_scalars[1] = loss;
        out_scalars[2] = entropy;
        out_scalars[3] = expf(entropy);
    }
}

__global__ void vq_init_ws_kernel(unsigned* __restrict__ loss_hist)
{
    int t = blockIdx.x * blockDim.x + threadIdx.x;
    if (t < 1025) loss_hist[t] = 0u;   // [0]=loss accumulator, [1..1024]=counts
}

// ---------------------------------------------------------------------------
extern "C" void kernel_launch(void* const* d_in, const int* in_sizes, int n_in,
                              void* d_out, int out_size, void* d_ws, size_t ws_size,
                              hipStream_t stream)
{
    (void)in_sizes; (void)n_in; (void)out_size; (void)ws_size;

    const float* z  = (const float*)d_in[0];
    const float* cb = (const float*)d_in[1];

    float* out         = (float*)d_out;
    float* out_q       = out;                          // 8,388,608
    float* out_idx     = out + OUT_Q_ELEMS;            // 131,072
    float* out_scalars = out + OUT_Q_ELEMS + NPAIRS;   // 4

    char* ws = (char*)d_ws;
    int*      idx_ws    = (int*)(ws + 0);              // 524288 B
    unsigned* loss_hist = (unsigned*)(ws + 524288);    // 4100 B
    float*    loss_acc  = (float*)loss_hist;
    unsigned* counts    = loss_hist + 1;
    float*    c2ws      = (float*)(ws + 532480);       // 131072 B
    unsigned* cbh32     = (unsigned*)(ws + 663552);    // 4 MB (bf16 hi, permuted)
    unsigned* cbl32     = (unsigned*)(ws + 663552 + 4194304); // 4 MB (bf16 lo)

    hipLaunchKernelGGL(vq_init_ws_kernel, dim3(2), dim3(1024), 0, stream, loss_hist);
    hipLaunchKernelGGL(vq_prep_kernel, dim3(NROWS / 8), dim3(256), 0, stream,
                       cb, cbh32, cbl32, c2ws);
    hipLaunchKernelGGL(vq_argmin_kernel, dim3(32, 32), dim3(256), 0, stream,
                       z, (const float4*)cbh32, (const float4*)cbl32, c2ws,
                       idx_ws, out_idx);
    hipLaunchKernelGGL(vq_gather_kernel, dim3(NPAIRS / 4), dim3(256), 0, stream,
                       z, cb, idx_ws, out_q, loss_acc, counts);
    hipLaunchKernelGGL(vq_finalize_kernel, dim3(1), dim3(1024), 0, stream,
                       counts, loss_acc, out_scalars);
}
